// NystromAttention_35433480192027
// MI455X (gfx1250) — matrix operand order repaired and checked by reading, not compiled, not run on hardware
//
#include <hip/hip_runtime.h>
#include <math.h>

// Problem constants (match reference)
#define BATCH  4
#define SEQ    4096
#define DMODEL 1024
#define NHEADS 16
#define HD     64
#define NLM    128
#define BH     (BATCH*NHEADS)   // 64
#define SEG    (SEQ/NLM)        // 32

typedef unsigned short u16;
typedef __attribute__((ext_vector_type(16))) __bf16        v16bf;
typedef __attribute__((ext_vector_type(8)))  float         v8f;
typedef __attribute__((ext_vector_type(2)))  float         v2f;
typedef __attribute__((ext_vector_type(4)))  float         v4f;
typedef __attribute__((ext_vector_type(4)))  unsigned int  u32x4;

static __device__ __forceinline__ u16 f2bf(float f) {
    unsigned u = __builtin_bit_cast(unsigned, f);
    unsigned r = u + 0x7FFFu + ((u >> 16) & 1u);   // round-to-nearest-even
    return (u16)(r >> 16);
}
static __device__ __forceinline__ float bf2f(u16 v) {
    unsigned u = ((unsigned)v) << 16;
    return __builtin_bit_cast(float, u);
}

// WMMA fragment loads: register image == memory image -> 16B loads + bit_cast.
struct bpair { u32x4 lo, hi; };
// A 16x32 bf16 layout: per-lane 16 contiguous elements at [2*hi8..2*hi8+15]
static __device__ __forceinline__ v16bf fragA(const u16* row, int hi8) {
    bpair f;
    f.lo = *(const u32x4*)(row + 2 * hi8);
    f.hi = *(const u32x4*)(row + 2 * hi8 + 8);
    return __builtin_bit_cast(v16bf, f);
}
// B 32x16 bf16 layout: per-lane 16 contiguous elements (caller adds kB)
static __device__ __forceinline__ v16bf fragB(const u16* p) {
    bpair f;
    f.lo = *(const u32x4*)(p);
    f.hi = *(const u32x4*)(p + 8);
    return __builtin_bit_cast(v16bf, f);
}

// CDNA5 async global->LDS copy (ASYNCcnt-tracked). lds_off is the addrspace(3)
// byte offset (low 32 bits of a generic __shared__ pointer).
static __device__ __forceinline__ void async_ld_b128(unsigned lds_off, const u16* gptr) {
    asm volatile("global_load_async_to_lds_b128 %0, %1, off"
                 :: "v"(lds_off), "v"(gptr) : "memory");
}
static __device__ __forceinline__ void wait_async0() {
    asm volatile("s_wait_asynccnt 0" ::: "memory");
}

// ---------------------------------------------------------------------------
// 0) Bulk f32 -> bf16 conversion (bandwidth bound). n multiple of 8.
// ---------------------------------------------------------------------------
__global__ __launch_bounds__(256) void cvt_bf16_kernel(
    const float* __restrict__ in, u16* __restrict__ out, int n)
{
    int i = (blockIdx.x * 256 + threadIdx.x) * 8;
    if (i >= n) return;
    v4f x0 = *(const v4f*)(in + i);
    v4f x1 = *(const v4f*)(in + i + 4);
    struct h8 { u16 v[8]; } r;
#pragma unroll
    for (int j = 0; j < 4; ++j) { r.v[j] = f2bf(x0[j]); r.v[4 + j] = f2bf(x1[j]); }
    *(u32x4*)(out + i) = __builtin_bit_cast(u32x4, r);
}

// ---------------------------------------------------------------------------
// 1) Projection: out[b,h,l,d] = bf16( (x @ W^T + bias) * scale )
//    128x128 block (8 waves), each wave a 64x32 strip: 8 WMMA accumulators,
//    4 A-frags + 2 B-frags per 32-wide k-step. Unconditional prefetch.
// ---------------------------------------------------------------------------
__global__ __launch_bounds__(256) void proj_kernel(
    const u16* __restrict__ Xb, const u16* __restrict__ Wb,
    const float* __restrict__ bias, u16* __restrict__ out, float scale)
{
    const int lane = threadIdx.x & 31;
    const int wave = threadIdx.x >> 5;        // 0..7
    const int ln   = lane & 15;
    const int hi   = (lane & 16) >> 1;        // 0 or 8
    const int kB   = lane & 16;               // 0 or 16
    const int mbase = blockIdx.x * 128 + (wave >> 2) * 64;
    const int nbase = blockIdx.y * 128 + (wave & 3) * 32;

    v8f acc[4][2] = {};
    for (int kk = 0; kk < DMODEL; kk += 32) {
        v16bf a[4], b[2];
#pragma unroll
        for (int ti = 0; ti < 4; ++ti) {
            const u16* xr = Xb + (size_t)(mbase + 16*ti + ln) * DMODEL + kk;
            __builtin_prefetch(xr + 32, 0, 1);
            a[ti] = fragA(xr, hi);
        }
#pragma unroll
        for (int tj = 0; tj < 2; ++tj) {
            const u16* wr = Wb + (size_t)(nbase + 16*tj + ln) * DMODEL + kk + kB;
            __builtin_prefetch(wr + 32, 0, 1);
            b[tj] = fragB(wr);
        }
#pragma unroll
        for (int ti = 0; ti < 4; ++ti)
#pragma unroll
            for (int tj = 0; tj < 2; ++tj)
                acc[ti][tj] = __builtin_amdgcn_wmma_f32_16x16x32_bf16(
                    false, a[ti], false, b[tj], (short)0, acc[ti][tj], false, false);
    }
#pragma unroll
    for (int ti = 0; ti < 4; ++ti) {
#pragma unroll
        for (int tj = 0; tj < 2; ++tj) {
            int col = nbase + 16*tj + ln;
            float bv = bias[col];
            int h = col >> 6, d = col & 63;
#pragma unroll
            for (int v = 0; v < 8; ++v) {
                int row  = mbase + 16*ti + v + hi;
                int bidx = row >> 12, l = row & 4095;
                float val = (acc[ti][tj][v] + bv) * scale;
                out[(((size_t)bidx * NHEADS + h) * SEQ + l) * HD + d] = f2bf(val);
            }
        }
    }
}

// ---------------------------------------------------------------------------
// 2) Landmarks: mean over 32-token segments. [bh,4096,64] -> [bh,128,64]
// ---------------------------------------------------------------------------
__global__ void landmark_kernel(const u16* __restrict__ in, u16* __restrict__ out)
{
    int idx = blockIdx.x * blockDim.x + threadIdx.x;
    if (idx >= BH * NLM * HD) return;
    int d = idx & 63, m = (idx >> 6) & 127, bh = idx >> 13;
    const u16* p = in + ((size_t)bh * SEQ + m * SEG) * HD + d;
    float s = 0.f;
#pragma unroll
    for (int i = 0; i < SEG; ++i) s += bf2f(p[(size_t)i * HD]);
    out[idx] = f2bf(s * (1.0f / SEG));
}

// ---------------------------------------------------------------------------
// 3) kernel_2 = softmax(Qlm @ Klm^T), one block per (bh,row)
// ---------------------------------------------------------------------------
__global__ __launch_bounds__(128) void k2_kernel(
    const u16* __restrict__ Qlm, const u16* __restrict__ Klm, float* __restrict__ K2)
{
    int bh = blockIdx.x >> 7;
    int m  = blockIdx.x & 127;
    int n  = threadIdx.x;
    __shared__ float buf[128];
    const u16* q = Qlm + ((size_t)bh * NLM + m) * HD;
    const u16* k = Klm + ((size_t)bh * NLM + n) * HD;
    float dot = 0.f;
#pragma unroll
    for (int d = 0; d < HD; ++d) dot += bf2f(q[d]) * bf2f(k[d]);
    buf[n] = dot; __syncthreads();
    for (int off = 64; off > 0; off >>= 1) {
        if (n < off) buf[n] = fmaxf(buf[n], buf[n + off]);
        __syncthreads();
    }
    float mx = buf[0]; __syncthreads();
    float e = __expf(dot - mx);
    buf[n] = e; __syncthreads();
    for (int off = 64; off > 0; off >>= 1) {
        if (n < off) buf[n] += buf[n + off];
        __syncthreads();
    }
    K2[((size_t)bh * NLM + m) * NLM + n] = e / buf[0];
}

// ---------------------------------------------------------------------------
// 4) global max of column sums (positive values -> int-compare atomicMax)
// ---------------------------------------------------------------------------
__global__ void init_gmax(int* g) { if (threadIdx.x == 0 && blockIdx.x == 0) *g = 0; }

__global__ __launch_bounds__(128) void colsum_max_kernel(
    const float* __restrict__ K2, int* __restrict__ g)
{
    int bh = blockIdx.x, n = threadIdx.x;
    const float* p = K2 + (size_t)bh * NLM * NLM + n;
    float s = 0.f;
    for (int m = 0; m < NLM; ++m) s += p[(size_t)m * NLM];
    atomicMax(g, __float_as_int(s));
}

__global__ void v0_kernel(const float* __restrict__ K2, const int* __restrict__ g,
                          float* __restrict__ V)
{
    int idx = blockIdx.x * blockDim.x + threadIdx.x;
    if (idx >= BH * NLM * NLM) return;
    int j = idx & 127, i = (idx >> 7) & 127, bh = idx >> 14;
    float gm = __int_as_float(*g);
    V[idx] = K2[((size_t)bh * NLM + j) * NLM + i] / gm;   // transpose / gmax
}

// ---------------------------------------------------------------------------
// 5) Newton-Schulz step (f32 WMMA 16x16x4): C = diag*I + sc*(A @ B'),
//    B' = bt ? (7I - B) : B (fused at LDS staging). 8 waves, 16-row strips.
// ---------------------------------------------------------------------------
__global__ __launch_bounds__(256) void nsmm_kernel(
    const float* __restrict__ A, const float* __restrict__ B,
    float* __restrict__ C, float diag, float sc, int bt)
{
    int bh   = blockIdx.x;
    int lane = threadIdx.x & 31, w = threadIdx.x >> 5;
    int ln   = lane & 15;
    int hi   = (lane & 16) >> 1;   // C/D row offset
    int kadd = (lane & 16) >> 3;   // 0 or 2 (f32 A/B k-group)
    const float* Ab = A + (size_t)bh * NLM * NLM;
    const float* Bb = B + (size_t)bh * NLM * NLM;
    float*       Cb = C + (size_t)bh * NLM * NLM;
    int mrow = w * 16 + ln;
    int col0 = ln;

    __shared__ float sB[16 * NLM];   // 8KB staged B panel

    v8f acc[8] = {};
    for (int kk = 0; kk < NLM; kk += 16) {
        __syncthreads();
#pragma unroll
        for (int i = threadIdx.x; i < 16 * NLM; i += 256) {
            int kr = kk + (i >> 7), c = i & 127;
            float bv = Bb[(size_t)kr * NLM + c];
            if (bt) bv = (kr == c ? 7.0f : 0.0f) - bv;
            sB[i] = bv;
        }
        __syncthreads();
#pragma unroll
        for (int kb = 0; kb < 4; ++kb) {
            v2f a;
            a[0] = Ab[(size_t)mrow * NLM + kk + kb * 4 + kadd];
            a[1] = Ab[(size_t)mrow * NLM + kk + kb * 4 + kadd + 1];
#pragma unroll
            for (int t = 0; t < 8; ++t) {
                v2f b;
                b[0] = sB[(kb * 4 + kadd) * NLM + t * 16 + col0];
                b[1] = sB[(kb * 4 + kadd + 1) * NLM + t * 16 + col0];
                acc[t] = __builtin_amdgcn_wmma_f32_16x16x4_f32(
                    false, a, false, b, (short)0, acc[t], false, false);
            }
        }
    }
#pragma unroll
    for (int t = 0; t < 8; ++t) {
        int col = t * 16 + col0;
#pragma unroll
        for (int v = 0; v < 8; ++v) {
            int row = w * 16 + v + hi;
            Cb[(size_t)row * NLM + col] = sc * acc[t][v] + (row == col ? diag : 0.0f);
        }
    }
}

// ---------------------------------------------------------------------------
// 6) F = softmax(Qlm @ K^T) @ V  (flash-style, online softmax over l).
//    K chunk staged via CDNA5 async global->LDS (ASYNCcnt); V chunk staged
//    manually into a transposed, bank-safe LDS tile.
// ---------------------------------------------------------------------------
#define VPAD 40   // row stride (u16): 16B-aligned rows, conflict-free banks
__global__ __launch_bounds__(128) void k3v_kernel(
    const u16* __restrict__ Qlm, const u16* __restrict__ Kh,
    const u16* __restrict__ Vh, float* __restrict__ F)
{
    int bh   = blockIdx.x;
    int lane = threadIdx.x & 31, w = threadIdx.x >> 5;
    int ln   = lane & 15, hi = (lane & 16) >> 1, kB = lane & 16;
    int mbase = blockIdx.y * 64 + w * 16;

    __shared__ u16   sK[32 * HD];            // 4KB row-major K chunk (async-staged)
    __shared__ u16   sVt[HD][VPAD];          // transposed V chunk [d][token]
    __shared__ float sSc[4][16][32];
    __shared__ u16   sPb[4][16][32];
    __shared__ float sMax[4][16], sSum[4][16], sCorr[4][16];

    // Qlm A-fragments (constant over the l loop)
    v16bf aq[2];
    const u16* qb = Qlm + ((size_t)bh * NLM + mbase + ln) * HD;
#pragma unroll
    for (int ks = 0; ks < 2; ++ks) aq[ks] = fragA(qb + ks * 32, hi);

    if (lane < 16) { sMax[w][lane] = -1e30f; sSum[w][lane] = 0.f; }

    v8f O[4] = {};
    for (int lb = 0; lb < SEQ; lb += 32) {
        __syncthreads();   // previous iteration's LDS readers done
        {   // async-stage K[lb..lb+31][0..63] row-major: 128 threads x 32B
            int t = threadIdx.x;
            const u16* gp = Kh + (size_t)bh * SEQ * HD + (size_t)lb * HD + t * 32;
            unsigned loff = (unsigned)(size_t)(void*)&sK[t * 32];
            async_ld_b128(loff, gp);
            async_ld_b128(loff + 32, gp + 16);
        }
        {   // manual transpose-stage of V[lb..lb+31][0..63]
            int t   = threadIdx.x;
            int tok = t >> 2;
            int d0  = (t & 3) * 16;
            const u16* vp = Vh + ((size_t)bh * SEQ + lb + tok) * HD + d0;
            union { u32x4 q; u16 s[8]; } a0, a1;
            a0.q = *(const u32x4*)vp;
            a1.q = *(const u32x4*)(vp + 8);
#pragma unroll
            for (int i = 0; i < 8; ++i) sVt[d0 + i][tok] = a0.s[i];
#pragma unroll
            for (int i = 0; i < 8; ++i) sVt[d0 + 8 + i][tok] = a1.s[i];
        }
        wait_async0();
        __syncthreads();   // sK (async) + sVt staged by all threads

        // scores: 16 rows x 32 l-cols = two 16x16 tiles (B-frags from LDS)
        v8f s[2] = {};
#pragma unroll
        for (int t = 0; t < 2; ++t) {
#pragma unroll
            for (int ks = 0; ks < 2; ++ks) {
                const u16* kr = sK + (16*t + ln) * HD + ks * 32 + kB;
                v16bf b = fragB(kr);
                s[t] = __builtin_amdgcn_wmma_f32_16x16x32_bf16(
                    false, aq[ks], false, b, (short)0, s[t], false, false);
            }
        }
#pragma unroll
        for (int t = 0; t < 2; ++t)
#pragma unroll
            for (int v = 0; v < 8; ++v)
                sSc[w][v + hi][t * 16 + ln] = s[t][v];
        // online softmax, one row per lane (lanes 0..15)
        if (lane < 16) {
            int r = lane;
            float cm = -1e30f;
#pragma unroll
            for (int c = 0; c < 32; ++c) cm = fmaxf(cm, sSc[w][r][c]);
            float om = sMax[w][r];
            float nm = fmaxf(om, cm);
            float corr = __expf(om - nm);
            float ssum = sSum[w][r] * corr;
#pragma unroll
            for (int c = 0; c < 32; ++c) {
                float e = __expf(sSc[w][r][c] - nm);
                ssum += e;
                sPb[w][r][c] = f2bf(e);
            }
            sMax[w][r] = nm; sSum[w][r] = ssum; sCorr[w][r] = corr;
        }
        // O correction + O += P @ V
        v16bf ap = fragA(&sPb[w][ln][0], hi);
#pragma unroll
        for (int td = 0; td < 4; ++td) {
#pragma unroll
            for (int v = 0; v < 8; ++v) O[td][v] *= sCorr[w][v + hi];
            v16bf bv = fragB(&sVt[td * 16 + ln][kB]);
            O[td] = __builtin_amdgcn_wmma_f32_16x16x32_bf16(
                false, ap, false, bv, (short)0, O[td], false, false);
        }
    }
#pragma unroll
    for (int td = 0; td < 4; ++td)
#pragma unroll
        for (int v = 0; v < 8; ++v) {
            int r = v + hi;
            F[((size_t)bh * NLM + mbase + r) * HD + td * 16 + ln] = O[td][v] / sSum[w][r];
        }
}

// ---------------------------------------------------------------------------
// 7) W2^T = (inv(kernel_2) @ F)^T : stored [bh][d][m] bf16 (contiguous B-frags)
// ---------------------------------------------------------------------------
__global__ __launch_bounds__(256) void w2_kernel(
    const float* __restrict__ Vns, const float* __restrict__ F, u16* __restrict__ W2t)
{
    int idx = blockIdx.x * 256 + threadIdx.x;
    if (idx >= BH * NLM * HD) return;
    int m = idx & 127, d = (idx >> 7) & 63, bh = idx >> 13;
    const float* vr = Vns + ((size_t)bh * NLM + m) * NLM;
    const float* fc = F + (size_t)bh * NLM * HD + d;
    float s = 0.f;
#pragma unroll 4
    for (int j = 0; j < NLM; ++j) s += vr[j] * fc[(size_t)j * HD];
    W2t[((size_t)bh * HD + d) * NLM + m] = f2bf(s);
}

// ---------------------------------------------------------------------------
// 8) out = softmax(Q @ Klm^T) @ W2, fused per 16-row wave strip.
// ---------------------------------------------------------------------------
__global__ __launch_bounds__(128) void final_kernel(
    const u16* __restrict__ Qh, const u16* __restrict__ Klm,
    const u16* __restrict__ W2t, float* __restrict__ out)
{
    int bh   = blockIdx.x;
    int lane = threadIdx.x & 31, w = threadIdx.x >> 5;
    int ln   = lane & 15, hi = (lane & 16) >> 1, kB = lane & 16;
    int lbase = blockIdx.y * 64 + w * 16;

    __shared__ float sSc[4][16][NLM];    // 32 KB raw scores
    __shared__ u16   sPbF[4][16][NLM];   // 16 KB bf16 exp(scores)
    __shared__ float sSumF[4][16];

    v16bf aq[2];
    const u16* qb = Qh + ((size_t)bh * SEQ + lbase + ln) * HD;
#pragma unroll
    for (int ks = 0; ks < 2; ++ks) aq[ks] = fragA(qb + ks * 32, hi);

#pragma unroll
    for (int t = 0; t < 8; ++t) {
        v8f s = {};
#pragma unroll
        for (int ks = 0; ks < 2; ++ks) {
            const u16* kr = Klm + ((size_t)bh * NLM + 16*t + ln) * HD + ks * 32 + kB;
            v16bf b = fragB(kr);
            s = __builtin_amdgcn_wmma_f32_16x16x32_bf16(
                false, aq[ks], false, b, (short)0, s, false, false);
        }
#pragma unroll
        for (int v = 0; v < 8; ++v) sSc[w][v + hi][t * 16 + ln] = s[v];
    }
    if (lane < 16) {
        int r = lane;
        float mx = -1e30f;
        for (int c = 0; c < NLM; ++c) mx = fmaxf(mx, sSc[w][r][c]);
        float ss = 0.f;
        for (int c = 0; c < NLM; ++c) {
            float e = __expf(sSc[w][r][c] - mx);
            sPbF[w][r][c] = f2bf(e);
            ss += e;
        }
        sSumF[w][r] = ss;
    }
    v8f O[4] = {};
#pragma unroll
    for (int ks = 0; ks < 4; ++ks) {
        v16bf ap = fragA(&sPbF[w][ln][ks * 32], hi);
#pragma unroll
        for (int td = 0; td < 4; ++td) {
            const u16* bp = W2t + ((size_t)bh * HD + td * 16 + ln) * NLM + ks * 32 + kB;
            v16bf b = fragB(bp);
            O[td] = __builtin_amdgcn_wmma_f32_16x16x32_bf16(
                false, ap, false, b, (short)0, O[td], false, false);
        }
    }
    int b_ = bh >> 4, h = bh & 15;
#pragma unroll
    for (int td = 0; td < 4; ++td)
#pragma unroll
        for (int v = 0; v < 8; ++v) {
            int l = lbase + v + hi;
            float inv = 1.0f / sSumF[w][v + hi];
            out[((size_t)b_ * SEQ + l) * DMODEL + h * HD + td * 16 + ln] = O[td][v] * inv;
        }
}

// ---------------------------------------------------------------------------
extern "C" void kernel_launch(void* const* d_in, const int* in_sizes, int n_in,
                              void* d_out, int out_size, void* d_ws, size_t ws_size,
                              hipStream_t stream)
{
    const float* x  = (const float*)d_in[0];
    const float* Wq = (const float*)d_in[1];
    const float* bq = (const float*)d_in[2];
    const float* Wk = (const float*)d_in[3];
    const float* bk = (const float*)d_in[4];
    const float* Wv = (const float*)d_in[5];
    const float* bv = (const float*)d_in[6];
    float* out = (float*)d_out;

    char* ws = (char*)d_ws;
    size_t off = 0;
    auto alloc = [&](size_t bytes) {
        void* p = ws + off;
        off += (bytes + 255) & ~(size_t)255;
        return p;
    };
    u16*   xbf  = (u16*)  alloc((size_t)BATCH * SEQ * DMODEL * sizeof(u16));
    u16*   Wqb  = (u16*)  alloc((size_t)DMODEL * DMODEL * sizeof(u16));
    u16*   Wkb  = (u16*)  alloc((size_t)DMODEL * DMODEL * sizeof(u16));
    u16*   Wvb  = (u16*)  alloc((size_t)DMODEL * DMODEL * sizeof(u16));
    u16*   Qh   = (u16*)  alloc((size_t)BH * SEQ * HD * sizeof(u16));
    u16*   Kh   = (u16*)  alloc((size_t)BH * SEQ * HD * sizeof(u16));
    u16*   Vh   = (u16*)  alloc((size_t)BH * SEQ * HD * sizeof(u16));
    u16*   Qlm  = (u16*)  alloc((size_t)BH * NLM * HD * sizeof(u16));
    u16*   Klm  = (u16*)  alloc((size_t)BH * NLM * HD * sizeof(u16));
    float* K2   = (float*)alloc((size_t)BH * NLM * NLM * sizeof(float));
    float* Vns0 = (float*)alloc((size_t)BH * NLM * NLM * sizeof(float));
    float* Vns1 = (float*)alloc((size_t)BH * NLM * NLM * sizeof(float));
    float* KV   = (float*)alloc((size_t)BH * NLM * NLM * sizeof(float));
    float* Tm   = (float*)alloc((size_t)BH * NLM * NLM * sizeof(float));
    float* Tm2  = (float*)alloc((size_t)BH * NLM * NLM * sizeof(float));
    float* F    = (float*)alloc((size_t)BH * NLM * HD * sizeof(float));
    u16*   W2t  = (u16*)  alloc((size_t)BH * NLM * HD * sizeof(u16));
    int*   gmax = (int*)  alloc(sizeof(int));

    const float scaleQK = 0.3535533905932738f;  // 64^-0.25

    // 0) bulk f32 -> bf16
    int nx = BATCH * SEQ * DMODEL, nw = DMODEL * DMODEL;
    cvt_bf16_kernel<<<(nx / 8 + 255) / 256, 256, 0, stream>>>(x,  xbf, nx);
    cvt_bf16_kernel<<<(nw / 8 + 255) / 256, 256, 0, stream>>>(Wq, Wqb, nw);
    cvt_bf16_kernel<<<(nw / 8 + 255) / 256, 256, 0, stream>>>(Wk, Wkb, nw);
    cvt_bf16_kernel<<<(nw / 8 + 255) / 256, 256, 0, stream>>>(Wv, Wvb, nw);

    // 1) projections (128x128 blocks)
    dim3 pg(SEQ * BATCH / 128, DMODEL / 128);
    proj_kernel<<<pg, 256, 0, stream>>>(xbf, Wqb, bq, Qh, scaleQK);
    proj_kernel<<<pg, 256, 0, stream>>>(xbf, Wkb, bk, Kh, scaleQK);
    proj_kernel<<<pg, 256, 0, stream>>>(xbf, Wvb, bv, Vh, 1.0f);

    // 2) landmarks
    int lmN = BH * NLM * HD;
    landmark_kernel<<<(lmN + 255) / 256, 256, 0, stream>>>(Qh, Qlm);
    landmark_kernel<<<(lmN + 255) / 256, 256, 0, stream>>>(Kh, Klm);

    // 3) kernel_2 softmax
    k2_kernel<<<BH * NLM, 128, 0, stream>>>(Qlm, Klm, K2);

    // 4) Newton-Schulz init: V0 = K2^T / max(colsum(K2))
    init_gmax<<<1, 1, 0, stream>>>(gmax);
    colsum_max_kernel<<<BH, 128, 0, stream>>>(K2, gmax);
    int v0N = BH * NLM * NLM;
    v0_kernel<<<(v0N + 255) / 256, 256, 0, stream>>>(K2, gmax, Vns0);

    // 5) 6 Newton-Schulz iterations (f32 WMMA)
    for (int it = 0; it < 6; ++it) {
        float* Vin  = (it & 1) ? Vns1 : Vns0;
        float* Vout = (it & 1) ? Vns0 : Vns1;
        nsmm_kernel<<<BH, 256, 0, stream>>>(K2, Vin, KV, 0.0f,  1.0f,  0);   // KV = K2@V
        nsmm_kernel<<<BH, 256, 0, stream>>>(KV, KV, Tm, 15.0f, -1.0f,  1);   // mid = 15I - KV@(7I-KV)
        nsmm_kernel<<<BH, 256, 0, stream>>>(KV, Tm, Tm2, 13.0f, -1.0f, 0);   // outer = 13I - KV@mid
        nsmm_kernel<<<BH, 256, 0, stream>>>(Vin, Tm2, Vout, 0.0f, 0.25f, 0); // V = 0.25*V@outer
    }
    // after 6 iterations the inverse is in Vns0

    // 6) F = softmax(Qlm @ K^T) @ V
    dim3 kg(BH, 2);
    k3v_kernel<<<kg, 128, 0, stream>>>(Qlm, Kh, Vh, F);

    // 7) W2^T = inv(K2) @ F
    int w2N = BH * NLM * HD;
    w2_kernel<<<(w2N + 255) / 256, 256, 0, stream>>>(Vns0, F, W2t);

    // 8) out = softmax(Q @ Klm^T) @ W2
    dim3 fg(BH, SEQ / 64);
    final_kernel<<<fg, 128, 0, stream>>>(Qh, Klm, W2t, out);
}
